// DeepSeekV3LatentAttention_64501818851863
// MI455X (gfx1250) — matrix operand order repaired and checked by reading, not compiled, not run on hardware
//
#include <hip/hip_runtime.h>
#include <math.h>

#define B_   2
#define S_   2048
#define DIN  2048
#define H_   16
#define RD_  64
#define L_   512
#define HD_  128
#define NQ   3072   // D_OUT + H*RD

typedef _Float16 v16h __attribute__((ext_vector_type(16)));
typedef _Float16 h8   __attribute__((ext_vector_type(8)));
typedef float    v8f  __attribute__((ext_vector_type(8)));
typedef int      i32x4 __attribute__((ext_vector_type(4)));

#if __has_builtin(__builtin_amdgcn_global_load_async_to_lds_b128) && \
    __has_builtin(__builtin_amdgcn_s_wait_asynccnt)
#define HAVE_ASYNC_LDS 1
#endif

// 16-byte copy global -> LDS (async path if available, else load+ds_store)
__device__ __forceinline__ void copy16_g2l(const _Float16* g, _Float16* l) {
#ifdef HAVE_ASYNC_LDS
  __builtin_amdgcn_global_load_async_to_lds_b128(
      (__attribute__((address_space(1))) i32x4*)g,
      (__attribute__((address_space(3))) i32x4*)l, 0, 0);
#else
  *(h8*)l = *(const h8*)g;
#endif
}
// wait until at most `n` async copies outstanding (loads complete in order)
__device__ __forceinline__ void wait_g2l_le4() {
#ifdef HAVE_ASYNC_LDS
  __builtin_amdgcn_s_wait_asynccnt(4);
#endif
}
__device__ __forceinline__ void wait_g2l_0() {
#ifdef HAVE_ASYNC_LDS
  __builtin_amdgcn_s_wait_asynccnt(0);
#endif
}

// ---------------------------------------------------------------- WMMA core
__device__ __forceinline__ v8f wmma_fma(v16h a, v16h b, v8f c) {
  return __builtin_amdgcn_wmma_f32_16x16x32_f16(
      false, a, false, b, (short)0, c, false, false);
}

// A fragment from an LDS row (K contiguous): elems 0..7 = K[8h..8h+7],
// elems 8..15 = K[16+8h .. 16+8h+7]
__device__ __forceinline__ v16h a_frag_lds(const _Float16* row, int hh) {
  h8 lo = *(const h8*)(row + 8 * hh);
  h8 hi = *(const h8*)(row + 16 + 8 * hh);
  v16h r;
#pragma unroll
  for (int i = 0; i < 8; ++i) { r[i] = lo[i]; r[i + 8] = hi[i]; }
  return r;
}
// B fragment from an LDS row: elems 0..7 = K[8h..8h+7], elems 8..15 = K[16+8h .. 16+8h+7]
__device__ __forceinline__ v16h b_frag_lds(const _Float16* row, int hh) {
  h8 lo = *(const h8*)(row + 8 * hh);
  h8 hi = *(const h8*)(row + 16 + 8 * hh);
  v16h r;
#pragma unroll
  for (int i = 0; i < 8; ++i) { r[i] = lo[i]; r[i + 8] = hi[i]; }
  return r;
}

// D tile store: lanes 0-15 -> N=lane, M=m0+r ; lanes 16-31 -> N=lane-16, M=m0+8+r
template <typename OutT, bool ACC>
__device__ __forceinline__ void store_tile(OutT* __restrict__ C, int ldc,
                                           int m0, int n0, int lane, v8f d) {
  const int n  = n0 + (lane & 15);
  const int mb = m0 + 8 * (lane >> 4);
  OutT* p = C + (size_t)mb * ldc + n;
  if (ACC) {
    float prev[8];
#pragma unroll
    for (int r = 0; r < 8; ++r) prev[r] = (float)p[(size_t)r * ldc];
#pragma unroll
    for (int r = 0; r < 8; ++r) p[(size_t)r * ldc] = (OutT)(prev[r] + d[r]);
  } else {
#pragma unroll
    for (int r = 0; r < 8; ++r) p[(size_t)r * ldc] = (OutT)d[r];
  }
}

// C[M,N] (+)= A[M,K] * B[N,K]^T ; fp16 operands, fp32 accum.
// 256 threads = 8 waves (2x4); block tile 128x128, wave tile 64x32, K step 32.
// Double-buffered LDS staging via async global->LDS copies.
template <typename OutT, bool ACC>
__global__ __launch_bounds__(256) void gemm_wmma_kernel(
    const _Float16* __restrict__ A, const _Float16* __restrict__ Bm,
    OutT* __restrict__ C, int M, int N, int K, int lda, int ldb, int ldc) {
  constexpr int LSTR = 40;  // 32 + 8 pad halves (80B row) -> conflict-free frags
  __shared__ __align__(16) _Float16 Alds[2][128 * LSTR];
  __shared__ __align__(16) _Float16 Blds[2][128 * LSTR];

  const int tid  = threadIdx.x;
  const int lane = tid & 31;
  const int wave = tid >> 5;
  const int wm = wave >> 2, wn = wave & 3;   // wave tile: rows wm*64, cols wn*32
  const int m_blk = blockIdx.y * 128;
  const int n_blk = blockIdx.x * 128;

  // staging: each thread copies 2 A chunks + 2 B chunks (rows r and r+64)
  const int row4 = tid >> 2;                 // 0..63
  const int seg  = (tid & 3) * 8;            // half offset 0/8/16/24
  const _Float16* ga0 = A  + (size_t)(m_blk + row4) * lda + seg;
  const _Float16* ga1 = ga0 + (size_t)64 * lda;
  const _Float16* gb0 = Bm + (size_t)(n_blk + row4) * ldb + seg;
  const _Float16* gb1 = gb0 + (size_t)64 * ldb;
  const int lofs = row4 * LSTR + seg;

  auto stage = [&](int buf, int k) {
    copy16_g2l(ga0 + k, &Alds[buf][lofs]);
    copy16_g2l(ga1 + k, &Alds[buf][lofs + 64 * LSTR]);
    copy16_g2l(gb0 + k, &Blds[buf][lofs]);
    copy16_g2l(gb1 + k, &Blds[buf][lofs + 64 * LSTR]);
  };

  const int hh = lane >> 4;
  const int arow = wm * 64 + (lane & 15);
  const int brow = wn * 32 + (lane & 15);

  v8f acc[4][2] = {};
  stage(0, 0);
  for (int k = 0; k < K; k += 32) {
    const int buf = (k >> 5) & 1;
    if (k + 32 < K) {          // prefetch next tile into other buffer first
      stage(buf ^ 1, k + 32);
      wait_g2l_le4();          // current tile's 4 copies are done (in-order)
    } else {
      wait_g2l_0();
    }
    __syncthreads();

    const _Float16* pa = &Alds[buf][arow * LSTR];
    const _Float16* pb = &Blds[buf][brow * LSTR];
    v16h b0 = b_frag_lds(pb, hh);
    v16h b1 = b_frag_lds(pb + 16 * LSTR, hh);
#pragma unroll
    for (int i = 0; i < 4; ++i) {
      v16h a = a_frag_lds(pa + i * 16 * LSTR, hh);
      acc[i][0] = wmma_fma(a, b0, acc[i][0]);
      acc[i][1] = wmma_fma(a, b1, acc[i][1]);
    }
    __syncthreads();           // protect buffer being restaged next iteration
  }

  const int m0 = m_blk + wm * 64;
  const int n0 = n_blk + wn * 32;
#pragma unroll
  for (int i = 0; i < 4; ++i) {
    store_tile<OutT, ACC>(C, ldc, m0 + i * 16, n0,      lane, acc[i][0]);
    store_tile<OutT, ACC>(C, ldc, m0 + i * 16, n0 + 16, lane, acc[i][1]);
  }
}

// ---------------------------------------------------------------- helpers
__global__ __launch_bounds__(256) void cvt_f32_f16_kernel(
    const float* __restrict__ src, _Float16* __restrict__ dst, size_t n) {
  size_t i = (size_t)blockIdx.x * 256 + threadIdx.x;
  if (i < n) dst[i] = (_Float16)src[i];
}

// W_UK [H*HD, L] -> [H][L][HD] fp16 (K(=HD)-contiguous B operand)
__global__ __launch_bounds__(256) void transpose_wuk_kernel(
    const float* __restrict__ wuk, _Float16* __restrict__ out) {
  size_t i = (size_t)blockIdx.x * 256 + threadIdx.x;  // i -> [h][l][d]
  if (i >= (size_t)H_ * L_ * HD_) return;
  int d = (int)(i % HD_);
  size_t t = i / HD_;
  int l = (int)(t % L_);
  int h = (int)(t / L_);
  out[i] = (_Float16)wuk[((size_t)(h * HD_ + d)) * L_ + l];
}

// RMSNorm rows of dkv [B*S, L]; emit c_kv fp16 [B,S,L] and c_kvT fp16 [B,L,S]
__global__ __launch_bounds__(256) void rmsnorm_kernel(
    const float* __restrict__ dkv, const float* __restrict__ w,
    _Float16* __restrict__ ckv, _Float16* __restrict__ ckvT) {
  const int row = blockIdx.x;           // b*S + t
  const int b = row / S_, t = row % S_;
  const float* x = dkv + (size_t)row * L_;
  __shared__ float red[256];
  float ss = 0.f;
  for (int i = threadIdx.x; i < L_; i += 256) { float v = x[i]; ss += v * v; }
  red[threadIdx.x] = ss; __syncthreads();
  for (int st = 128; st > 0; st >>= 1) {
    if (threadIdx.x < st) red[threadIdx.x] += red[threadIdx.x + st];
    __syncthreads();
  }
  const float inv = rsqrtf(red[0] / (float)L_ + 1e-6f);
  for (int i = threadIdx.x; i < L_; i += 256) {
    _Float16 hv = (_Float16)(x[i] * inv * w[i]);
    ckv[(size_t)row * L_ + i] = hv;
    ckvT[((size_t)b * L_ + i) * S_ + t] = hv;
  }
}

// Split q_total; apply GPT-NeoX RoPE to q_rope and k_rope; repack per-head fp16.
__global__ __launch_bounds__(256) void rope_pack_kernel(
    const float* __restrict__ qtot, const float* __restrict__ krope,
    const int* __restrict__ offset_p,
    _Float16* __restrict__ q_content, _Float16* __restrict__ q_rope,
    _Float16* __restrict__ k_rope) {
  const int row = blockIdx.x;           // b*S + s
  const int b = row / S_, s = row % S_;
  const float pos = (float)(s + offset_p[0]);
  const float* qrow = qtot + (size_t)row * NQ;
  const float* krow = krope + (size_t)row * (H_ * RD_);
  for (int i = threadIdx.x; i < H_ * HD_; i += 256) {
    int h = i >> 7, d = i & 127;
    q_content[(((size_t)b * H_ + h) * S_ + s) * HD_ + d] = (_Float16)qrow[i];
  }
  for (int i = threadIdx.x; i < H_ * (RD_ / 2); i += 256) {
    int h = i >> 5, j = i & 31;
    float ang = pos * powf(10000.f, -(float)j / 32.f);
    float c = cosf(ang), sn = sinf(ang);
    size_t base = (((size_t)b * H_ + h) * S_ + s) * RD_;
    float x1 = qrow[H_ * HD_ + h * RD_ + j];
    float x2 = qrow[H_ * HD_ + h * RD_ + 32 + j];
    q_rope[base + j]      = (_Float16)(x1 * c - x2 * sn);
    q_rope[base + 32 + j] = (_Float16)(x2 * c + x1 * sn);
    x1 = krow[h * RD_ + j];
    x2 = krow[h * RD_ + 32 + j];
    k_rope[base + j]      = (_Float16)(x1 * c - x2 * sn);
    k_rope[base + 32 + j] = (_Float16)(x2 * c + x1 * sn);
  }
}

// Causal masked softmax of one score row; scale = rsqrt(HD+RD); P out fp16.
__global__ __launch_bounds__(256) void softmax_kernel(
    const float* __restrict__ scores, _Float16* __restrict__ P,
    const int* __restrict__ offset_p) {
  const int s = blockIdx.x;
  const int valid = min(S_, s + offset_p[0] + 1);
  const float scale = rsqrtf((float)(HD_ + RD_));
  const float* row = scores + (size_t)s * S_;
  __shared__ float red[256];
  float mx = -INFINITY;
  for (int t = threadIdx.x; t < valid; t += 256) mx = fmaxf(mx, row[t] * scale);
  red[threadIdx.x] = mx; __syncthreads();
  for (int st = 128; st > 0; st >>= 1) {
    if (threadIdx.x < st) red[threadIdx.x] = fmaxf(red[threadIdx.x], red[threadIdx.x + st]);
    __syncthreads();
  }
  const float rowmax = red[0]; __syncthreads();
  float sum = 0.f;
  for (int t = threadIdx.x; t < valid; t += 256) sum += expf(row[t] * scale - rowmax);
  red[threadIdx.x] = sum; __syncthreads();
  for (int st = 128; st > 0; st >>= 1) {
    if (threadIdx.x < st) red[threadIdx.x] += red[threadIdx.x + st];
    __syncthreads();
  }
  const float inv = 1.f / red[0];
  for (int t = threadIdx.x; t < S_; t += 256) {
    float p = (t < valid) ? expf(row[t] * scale - rowmax) * inv : 0.f;
    P[(size_t)s * S_ + t] = (_Float16)p;
  }
}

// ---------------------------------------------------------------- launch
extern "C" void kernel_launch(void* const* d_in, const int* in_sizes, int n_in,
                              void* d_out, int out_size, void* d_ws, size_t ws_size,
                              hipStream_t stream) {
  const float* x       = (const float*)d_in[0];
  const float* W_DKV   = (const float*)d_in[1];
  const float* W_KRope = (const float*)d_in[2];
  const float* W_Q     = (const float*)d_in[3];
  const float* W_UK    = (const float*)d_in[4];
  const float* W_UV    = (const float*)d_in[5];
  const float* W_O     = (const float*)d_in[6];
  const float* kvnw    = (const float*)d_in[7];
  const int*   offset  = (const int*)d_in[8];
  float* out = (float*)d_out;

  char* wsp = (char*)d_ws; size_t off = 0;
  auto alloc = [&](size_t bytes) -> void* {
    off = (off + 255) & ~(size_t)255;
    void* p = wsp + off; off += bytes; return p;
  };
  const size_t BS = (size_t)B_ * S_;            // 4096 rows
  _Float16* x_h     = (_Float16*)alloc(BS * DIN * 2);
  _Float16* wdkv_h  = (_Float16*)alloc((size_t)L_ * DIN * 2);
  _Float16* wkr_h   = (_Float16*)alloc((size_t)H_ * RD_ * DIN * 2);
  _Float16* wq_h    = (_Float16*)alloc((size_t)NQ * DIN * 2);
  _Float16* wukt_h  = (_Float16*)alloc((size_t)H_ * L_ * HD_ * 2);
  _Float16* wuv_h   = (_Float16*)alloc((size_t)H_ * HD_ * L_ * 2);
  _Float16* wo_h    = (_Float16*)alloc((size_t)DIN * DIN * 2);
  float*    dkv_f   = (float*)   alloc(BS * L_ * 4);
  float*    krope_f = (float*)   alloc(BS * H_ * RD_ * 4);
  float*    qtot_f  = (float*)   alloc(BS * NQ * 4);
  _Float16* ckv_h   = (_Float16*)alloc(BS * L_ * 2);
  _Float16* ckvT_h  = (_Float16*)alloc(BS * L_ * 2);
  _Float16* qcont_h = (_Float16*)alloc(BS * H_ * HD_ * 2);
  _Float16* qrope_h = (_Float16*)alloc(BS * H_ * RD_ * 2);
  _Float16* krop_h  = (_Float16*)alloc(BS * H_ * RD_ * 2);
  _Float16* qabs_h  = (_Float16*)alloc((size_t)S_ * L_ * 2);   // per (b,h)
  float*    sc_f    = (float*)   alloc((size_t)S_ * S_ * 4);   // per (b,h)
  _Float16* P_h     = (_Float16*)alloc((size_t)S_ * S_ * 2);   // per (b,h)
  _Float16* clat_h  = (_Float16*)alloc((size_t)S_ * L_ * 2);   // per (b,h)
  _Float16* ctx_h   = (_Float16*)alloc(BS * DIN * 2);

  auto cvt = [&](const float* s, _Float16* d, size_t n) {
    cvt_f32_f16_kernel<<<(unsigned)((n + 255) / 256), 256, 0, stream>>>(s, d, n);
  };
  auto gemm_f32 = [&](const _Float16* A, const _Float16* Bm, float* C,
                      int M, int N, int K, int lda, int ldb, int ldc, bool acc) {
    dim3 grid(N / 128, M / 128);
    if (acc) gemm_wmma_kernel<float, true ><<<grid, 256, 0, stream>>>(A, Bm, C, M, N, K, lda, ldb, ldc);
    else     gemm_wmma_kernel<float, false><<<grid, 256, 0, stream>>>(A, Bm, C, M, N, K, lda, ldb, ldc);
  };
  auto gemm_f16 = [&](const _Float16* A, const _Float16* Bm, _Float16* C,
                      int M, int N, int K, int lda, int ldb, int ldc) {
    dim3 grid(N / 128, M / 128);
    gemm_wmma_kernel<_Float16, false><<<grid, 256, 0, stream>>>(A, Bm, C, M, N, K, lda, ldb, ldc);
  };

  // 1) convert inputs / weights to fp16
  cvt(x, x_h, BS * DIN);
  cvt(W_DKV, wdkv_h, (size_t)L_ * DIN);
  cvt(W_KRope, wkr_h, (size_t)H_ * RD_ * DIN);
  cvt(W_Q, wq_h, (size_t)NQ * DIN);
  cvt(W_UV, wuv_h, (size_t)H_ * HD_ * L_);
  cvt(W_O, wo_h, (size_t)DIN * DIN);
  transpose_wuk_kernel<<<(H_ * L_ * HD_) / 256, 256, 0, stream>>>(W_UK, wukt_h);

  // 2) input projections
  gemm_f32(x_h, wdkv_h, dkv_f,   (int)BS, L_,       DIN, DIN, DIN, L_,       false);
  gemm_f32(x_h, wkr_h,  krope_f, (int)BS, H_ * RD_, DIN, DIN, DIN, H_ * RD_, false);
  gemm_f32(x_h, wq_h,   qtot_f,  (int)BS, NQ,       DIN, DIN, DIN, NQ,       false);

  // 3) RMSNorm latent + RoPE repack
  rmsnorm_kernel<<<(unsigned)BS, 256, 0, stream>>>(dkv_f, kvnw, ckv_h, ckvT_h);
  rope_pack_kernel<<<(unsigned)BS, 256, 0, stream>>>(qtot_f, krope_f, offset,
                                                     qcont_h, qrope_h, krop_h);

  // 4) attention per (batch, head)
  for (int b = 0; b < B_; ++b) {
    for (int h = 0; h < H_; ++h) {
      const size_t bh = (size_t)b * H_ + h;
      // q_abs = q_content . W_UK_t[h]^T
      gemm_f16(qcont_h + bh * S_ * HD_, wukt_h + (size_t)h * L_ * HD_, qabs_h,
               S_, L_, HD_, HD_, HD_, L_);
      // scores = q_abs . c_kv^T + q_rope . k_rope^T
      gemm_f32(qabs_h, ckv_h + (size_t)b * S_ * L_, sc_f,
               S_, S_, L_, L_, L_, S_, false);
      gemm_f32(qrope_h + bh * S_ * RD_, krop_h + bh * S_ * RD_, sc_f,
               S_, S_, RD_, RD_, RD_, S_, true);
      // masked softmax
      softmax_kernel<<<S_, 256, 0, stream>>>(sc_f, P_h, offset);
      // ctx_lat = P . c_kv   (B operand = c_kvT, K=T contiguous)
      gemm_f16(P_h, ckvT_h + (size_t)b * L_ * S_, clat_h,
               S_, L_, S_, S_, S_, L_);
      // ctx[:, h*HD:(h+1)*HD] = ctx_lat . w_uv[h]^T
      gemm_f16(clat_h, wuv_h + (size_t)h * HD_ * L_,
               ctx_h + (size_t)b * S_ * DIN + (size_t)h * HD_,
               S_, HD_, L_, L_, L_, DIN);
    }
  }

  // 5) output projection
  gemm_f32(ctx_h, wo_h, out, (int)BS, DIN, DIN, DIN, DIN, DIN, false);
}
